// Attention_75110388073064
// MI455X (gfx1250) — matrix operand order repaired and checked by reading, not compiled, not run on hardware
//
#include <hip/hip_runtime.h>

#define NH 32
#define NKVH 8
#define HD 128
#define HIDDEN 4096
#define QKV_N 6144
#define T_TOK 1024
#define TOTAL_KV 3072

typedef __attribute__((ext_vector_type(16))) __bf16 v16bf;
typedef __attribute__((ext_vector_type(8))) float v8f;

union ABFrag { v16bf v; unsigned short h[16]; };

__device__ __forceinline__ unsigned short f2bf(float f) {
  unsigned int u = __float_as_uint(f);
  u += 0x7fffu + ((u >> 16) & 1u);   // round-to-nearest-even bf16
  return (unsigned short)(u >> 16);
}

__device__ __forceinline__ unsigned short ld_bf(const float* p, long i) { return f2bf(p[i]); }
__device__ __forceinline__ unsigned short ld_bf(const unsigned short* p, long i) { return p[i]; }

// A fragment: 16x32 (MxK), element (m,k) at src[m*ld + k]. Contiguous in k ->
// two global_load_b128 per lane when src is bf16.
// Contiguous-k layout used here (matches load_b): lanes 0-15 hold M=lane, K=0..15; lanes 16-31 K=16..31.
template <typename T>
__device__ __forceinline__ v16bf load_a(const T* src, int ld) {
  const int lane = threadIdx.x & 31;
  const int m = lane & 15;
  const int kb = (lane >> 4) << 4;   // 0 or 16
  ABFrag r;
  const long row = (long)m * ld;
#pragma unroll
  for (int i = 0; i < 4; ++i) {
    r.h[2 * i]         = ld_bf(src, row + kb + 2 * i);
    r.h[2 * i + 1]     = ld_bf(src, row + kb + 2 * i + 1);
    r.h[8 + 2 * i]     = ld_bf(src, row + 8 + kb + 2 * i);
    r.h[8 + 2 * i + 1] = ld_bf(src, row + 8 + kb + 2 * i + 1);
  }
  return r.v;
}

// B fragment: 32x16 (KxN), element (k,n) at src[k*sk + n*sn].
// With sk==1 (pre-transposed operand) this is 16 contiguous bf16 per lane.
// Contiguous-k layout used here (matches load_a): lanes 0-15 hold N=lane, K=0..15; lanes 16-31 K=16..31.
template <typename T>
__device__ __forceinline__ v16bf load_b(const T* src, long sk, long sn) {
  const int lane = threadIdx.x & 31;
  const int n = lane & 15;
  const int kb = (lane >> 4) << 4;   // 0 or 16
  ABFrag r;
#pragma unroll
  for (int i = 0; i < 8; ++i) {
    r.h[2 * i]     = ld_bf(src, (long)(kb + 2 * i) * sk + (long)n * sn);
    r.h[2 * i + 1] = ld_bf(src, (long)(kb + 2 * i + 1) * sk + (long)n * sn);
  }
  return r.v;
}

__device__ __forceinline__ v8f wmma_bf16(v16bf a, v16bf b, v8f c) {
  return __builtin_amdgcn_wmma_f32_16x16x32_bf16(false, a, false, b, (short)0, c, false, false);
}

// ---------------------------------------------------------------------------
// Elementwise f32 -> bf16 convert.
// ---------------------------------------------------------------------------
__global__ void cvt_kernel(const float* __restrict__ src, unsigned short* __restrict__ dst, long n) {
  long i = (long)blockIdx.x * blockDim.x + threadIdx.x;
  const long stride = (long)gridDim.x * blockDim.x;
  for (; i < n; i += stride) dst[i] = f2bf(src[i]);
}

// ---------------------------------------------------------------------------
// Tiled transpose + convert: dst(C x R) bf16 = src(R x C)^T. Block (32,8).
// ---------------------------------------------------------------------------
template <typename T>
__global__ void transpose_cvt_kernel(const T* __restrict__ src, unsigned short* __restrict__ dst,
                                     int R, int C) {
  __shared__ unsigned short tile[32][34];
  const int c0 = blockIdx.x * 32;
  const int r0 = blockIdx.y * 32;
  const int tx = threadIdx.x;
  const int ty = threadIdx.y;
#pragma unroll
  for (int i = 0; i < 4; ++i)
    tile[ty + i * 8][tx] = ld_bf(src, (long)(r0 + ty + i * 8) * C + c0 + tx);
  __syncthreads();
#pragma unroll
  for (int i = 0; i < 4; ++i)
    dst[(long)(c0 + ty + i * 8) * R + r0 + tx] = tile[tx][ty + i * 8];
}

// ---------------------------------------------------------------------------
// GEMM (TN): C(f32, MxN) = A(MxK, bf16 row-major) * Bt(NxK, bf16 row-major)^T.
// 8 waves/block, 32x64 per wave -> 64x256 block tile. All frag loads are
// contiguous b128; 8 WMMA per k-step per wave.
// ---------------------------------------------------------------------------
__global__ void __launch_bounds__(256) gemm_tn_kernel(const unsigned short* __restrict__ A,
                                                      const unsigned short* __restrict__ Bt,
                                                      float* __restrict__ C,
                                                      int M, int N, int K) {
  const int wave = threadIdx.x >> 5;
  const int m0 = blockIdx.x * 64 + (wave >> 2) * 32;
  const int n0 = blockIdx.y * 256 + (wave & 3) * 64;
  v8f acc[2][4] = {};
  for (int k0 = 0; k0 < K; k0 += 32) {
    v16bf a[2], b[4];
#pragma unroll
    for (int i = 0; i < 2; ++i)
      a[i] = load_a(A + (long)(m0 + 16 * i) * K + k0, K);
#pragma unroll
    for (int j = 0; j < 4; ++j)
      b[j] = load_b(Bt + (long)(n0 + 16 * j) * K + k0, 1L, (long)K);
#pragma unroll
    for (int i = 0; i < 2; ++i)
#pragma unroll
      for (int j = 0; j < 4; ++j)
        acc[i][j] = wmma_bf16(a[i], b[j], acc[i][j]);
  }
  const int lane = threadIdx.x & 31;
  const int n = lane & 15;
  const int mb = (lane >> 4) << 3;
#pragma unroll
  for (int i = 0; i < 2; ++i)
#pragma unroll
    for (int j = 0; j < 4; ++j)
#pragma unroll
      for (int r = 0; r < 8; ++r)
        C[(long)(m0 + 16 * i + mb + r) * N + n0 + 16 * j + n] = acc[i][j][r];
}

// ---------------------------------------------------------------------------
// RoPE on Q, convert to bf16. One thread per (t, h, pair).
// ---------------------------------------------------------------------------
__global__ void prep_q_kernel(const float* __restrict__ qkv, unsigned short* __restrict__ q_bf,
                              const int* __restrict__ seqstarts, const int* __restrict__ start_pos) {
  int idx = blockIdx.x * blockDim.x + threadIdx.x;
  if (idx >= T_TOK * NH * (HD / 2)) return;
  const int j = idx & 63;
  const int h = (idx >> 6) & 31;
  const int t = idx >> 11;
  int b = 0;
#pragma unroll
  for (int i = 1; i < 4; ++i) b = (t >= seqstarts[i]) ? i : b;
  const int pos = t - seqstarts[b] + start_pos[b];
  const float inv = __expf(-0.14391156831f * (float)j);  // 10000^(-j/64)
  float s, c;
  __sincosf((float)pos * inv, &s, &c);
  const float* src = qkv + (long)t * QKV_N + h * HD;
  const float x1 = src[2 * j], x2 = src[2 * j + 1];
  unsigned short* dst = q_bf + (long)t * HIDDEN + h * HD;
  dst[2 * j]     = f2bf(x1 * c - x2 * s);
  dst[2 * j + 1] = f2bf(x1 * s + x2 * c);
}

// ---------------------------------------------------------------------------
// Gather K/V per kv-token (RoPE new K), convert to bf16 (TOTAL_KV, NKVH, HD).
// ---------------------------------------------------------------------------
__global__ void prep_kv_kernel(const float* __restrict__ qkv, const float* __restrict__ kv_cache,
                               unsigned short* __restrict__ k_bf, unsigned short* __restrict__ v_bf,
                               const int* __restrict__ seqstarts, const int* __restrict__ kvstarts,
                               const int* __restrict__ cachestarts, const int* __restrict__ start_pos) {
  int idx = blockIdx.x * blockDim.x + threadIdx.x;
  if (idx >= TOTAL_KV * NKVH * (HD / 2)) return;
  const int j = idx & 63;
  const int h = (idx >> 6) & 7;
  const int t = idx >> 9;
  int b = 0;
#pragma unroll
  for (int i = 1; i < 4; ++i) b = (t >= kvstarts[i]) ? i : b;
  const int pkv = t - kvstarts[b];
  unsigned short* kd = k_bf + (long)t * (NKVH * HD) + h * HD;
  unsigned short* vd = v_bf + (long)t * (NKVH * HD) + h * HD;
  if (pkv >= start_pos[b]) {
    int nt = seqstarts[b] + pkv - start_pos[b];
    nt = nt < 0 ? 0 : (nt > T_TOK - 1 ? T_TOK - 1 : nt);
    const float* ks = qkv + (long)nt * QKV_N + HIDDEN + h * HD;
    const float* vs = qkv + (long)nt * QKV_N + HIDDEN + NKVH * HD + h * HD;
    const float inv = __expf(-0.14391156831f * (float)j);
    float s, c;
    __sincosf((float)pkv * inv, &s, &c);
    const float x1 = ks[2 * j], x2 = ks[2 * j + 1];
    kd[2 * j]     = f2bf(x1 * c - x2 * s);
    kd[2 * j + 1] = f2bf(x1 * s + x2 * c);
    vd[2 * j]     = f2bf(vs[2 * j]);
    vd[2 * j + 1] = f2bf(vs[2 * j + 1]);
  } else {
    const int ci = cachestarts[b] + pkv;
    const float* ks = kv_cache + (long)ci * (NKVH * HD) + h * HD;
    const float* vs = kv_cache + (long)8192 * NKVH * HD + (long)ci * (NKVH * HD) + h * HD;
    kd[2 * j]     = f2bf(ks[2 * j]);
    kd[2 * j + 1] = f2bf(ks[2 * j + 1]);
    vd[2 * j]     = f2bf(vs[2 * j]);
    vd[2 * j + 1] = f2bf(vs[2 * j + 1]);
  }
}

// ---------------------------------------------------------------------------
// Flash attention: one wave per (head, 16-row q tile); KV in tiles of 32,
// clipped to [min(lo), max(hi)] of the tile (causal + ragged early-exit).
// K is (t, kvh, d) bf16 (contiguous d); V is pre-transposed (kvh*d, t) bf16.
// ---------------------------------------------------------------------------
__global__ void __launch_bounds__(256) attn_kernel(const unsigned short* __restrict__ q_bf,
                                                   const unsigned short* __restrict__ k_bf,
                                                   const unsigned short* __restrict__ v_t,
                                                   unsigned short* __restrict__ attn_bf,
                                                   const int* __restrict__ seqstarts,
                                                   const int* __restrict__ kvstarts,
                                                   const int* __restrict__ start_pos) {
  __shared__ __align__(16) unsigned short p_lds[8][16 * 32];
  const int wave = threadIdx.x >> 5;
  const int lane = threadIdx.x & 31;
  const int gw = blockIdx.x * 8 + wave;
  const int h = gw >> 6;                 // 64 q-tiles per head
  const int q0 = (gw & 63) * 16;
  const int kvh = h >> 2;                // GQA: 4 q heads per kv head
  const int n = lane & 15;
  const int mb = (lane >> 4) << 3;

  // Per-row valid kv range [lo, hi] (inclusive): batch search done ONCE here.
  int lo[8], hi[8];
  int lomin, himax;
#pragma unroll
  for (int r = 0; r < 8; ++r) {
    const int t = q0 + mb + r;
    int b = 0;
#pragma unroll
    for (int i = 1; i < 4; ++i) b = (t >= seqstarts[i]) ? i : b;
    const int pos = t - seqstarts[b] + start_pos[b];
    lo[r] = kvstarts[b];
    hi[r] = lo[r] + pos;
    if (r == 0) { lomin = lo[0]; himax = hi[0]; }
    else {
      lomin = lo[r] < lomin ? lo[r] : lomin;
      himax = hi[r] > himax ? hi[r] : himax;
    }
  }
  // combine the two lane-halves (rows 0-7 vs 8-15), make scalar loop bounds
  {
    int o1 = __shfl_xor(lomin, 16, 32);
    lomin = o1 < lomin ? o1 : lomin;
    int o2 = __shfl_xor(himax, 16, 32);
    himax = o2 > himax ? o2 : himax;
  }
  const int kv_begin = __builtin_amdgcn_readfirstlane((lomin >> 5) << 5);
  const int kv_end = __builtin_amdgcn_readfirstlane(himax);

  v16bf qa[4];
#pragma unroll
  for (int kk = 0; kk < 4; ++kk)
    qa[kk] = load_a(q_bf + (long)q0 * HIDDEN + h * HD + kk * 32, HIDDEN);

  v8f o[8] = {};
  float m_r[8], l_r[8];
#pragma unroll
  for (int r = 0; r < 8; ++r) { m_r[r] = -3e38f; l_r[r] = 0.f; }

  const float scale = 0.08838834764831845f;  // 1/sqrt(128)

  for (int kv0 = kv_begin; kv0 <= kv_end; kv0 += 32) {
    // S = Q * K^T for two 16-col halves
    v8f s0 = {}, s1 = {};
#pragma unroll
    for (int kk = 0; kk < 4; ++kk) {
      v16bf kf0 = load_b(k_bf + (long)kv0 * (NKVH * HD) + kvh * HD + kk * 32, 1L, (long)(NKVH * HD));
      s0 = wmma_bf16(qa[kk], kf0, s0);
      v16bf kf1 = load_b(k_bf + (long)(kv0 + 16) * (NKVH * HD) + kvh * HD + kk * 32, 1L, (long)(NKVH * HD));
      s1 = wmma_bf16(qa[kk], kf1, s1);
    }
    const int col0 = kv0 + n, col1 = kv0 + 16 + n;

    float sv0[8], sv1[8], tmax[8];
    bool ok0[8], ok1[8];
#pragma unroll
    for (int r = 0; r < 8; ++r) {
      sv0[r] = s0[r] * scale;
      sv1[r] = s1[r] * scale;
      ok0[r] = (col0 >= lo[r]) && (col0 <= hi[r]);
      ok1[r] = (col1 >= lo[r]) && (col1 <= hi[r]);
      tmax[r] = fmaxf(ok0[r] ? sv0[r] : -3e38f, ok1[r] ? sv1[r] : -3e38f);
    }
#pragma unroll
    for (int off = 8; off >= 1; off >>= 1)
#pragma unroll
      for (int r = 0; r < 8; ++r) tmax[r] = fmaxf(tmax[r], __shfl_xor(tmax[r], off, 16));

    float sc[8], rsum[8];
#pragma unroll
    for (int r = 0; r < 8; ++r) {
      const float mnew = fmaxf(m_r[r], tmax[r]);
      sc[r] = __expf(m_r[r] - mnew);
      m_r[r] = mnew;
      const float e0 = ok0[r] ? __expf(sv0[r] - mnew) : 0.f;
      const float e1 = ok1[r] ? __expf(sv1[r] - mnew) : 0.f;
      p_lds[wave][(mb + r) * 32 + n]      = f2bf(e0);
      p_lds[wave][(mb + r) * 32 + 16 + n] = f2bf(e1);
      rsum[r] = e0 + e1;
    }
#pragma unroll
    for (int off = 8; off >= 1; off >>= 1)
#pragma unroll
      for (int r = 0; r < 8; ++r) rsum[r] += __shfl_xor(rsum[r], off, 16);
#pragma unroll
    for (int r = 0; r < 8; ++r) l_r[r] = l_r[r] * sc[r] + rsum[r];
#pragma unroll
    for (int dc = 0; dc < 8; ++dc)
#pragma unroll
      for (int r = 0; r < 8; ++r) o[dc][r] *= sc[r];

    asm volatile("s_wait_dscnt 0" ::: "memory");   // P store -> A-frag reload
    v16bf pa = load_a(&p_lds[wave][0], 32);
#pragma unroll
    for (int dc = 0; dc < 8; ++dc) {
      v16bf vf = load_b(v_t + (long)(kvh * HD + dc * 16) * TOTAL_KV + kv0, 1L, (long)TOTAL_KV);
      o[dc] = wmma_bf16(pa, vf, o[dc]);
    }
  }

#pragma unroll
  for (int r = 0; r < 8; ++r) {
    const float invl = 1.0f / l_r[r];
    unsigned short* dst = attn_bf + (long)(q0 + mb + r) * HIDDEN + h * HD;
#pragma unroll
    for (int dc = 0; dc < 8; ++dc) dst[dc * 16 + n] = f2bf(o[dc][r] * invl);
  }
}

// ---------------------------------------------------------------------------
extern "C" void kernel_launch(void* const* d_in, const int* in_sizes, int n_in,
                              void* d_out, int out_size, void* d_ws, size_t ws_size,
                              hipStream_t stream) {
  (void)in_sizes; (void)n_in; (void)out_size; (void)ws_size;
  const float* x         = (const float*)d_in[0];
  const float* wqkv      = (const float*)d_in[1];
  const float* wo        = (const float*)d_in[2];
  const float* kv_cache  = (const float*)d_in[3];
  const int* seqstarts   = (const int*)d_in[4];
  const int* kvstarts    = (const int*)d_in[5];
  const int* cachestarts = (const int*)d_in[6];
  const int* start_pos   = (const int*)d_in[7];

  char* ws = (char*)d_ws;
  float* qkv = (float*)ws;                        ws += (size_t)T_TOK * QKV_N * 4;
  unsigned short* x_bf = (unsigned short*)ws;     ws += (size_t)T_TOK * HIDDEN * 2;
  unsigned short* wqkv_t = (unsigned short*)ws;   ws += (size_t)HIDDEN * QKV_N * 2;
  unsigned short* wo_t = (unsigned short*)ws;     ws += (size_t)HIDDEN * HIDDEN * 2;
  unsigned short* q_bf = (unsigned short*)ws;     ws += (size_t)T_TOK * HIDDEN * 2;
  unsigned short* k_bf = (unsigned short*)ws;     ws += (size_t)TOTAL_KV * NKVH * HD * 2;
  unsigned short* v_bf = (unsigned short*)ws;     ws += (size_t)TOTAL_KV * NKVH * HD * 2;
  unsigned short* v_t = (unsigned short*)ws;      ws += (size_t)TOTAL_KV * NKVH * HD * 2;
  unsigned short* attn_bf = (unsigned short*)ws;  ws += (size_t)T_TOK * HIDDEN * 2;

  // 0) One-pass conversions/transposes -> pure-bf16 contiguous WMMA operands
  cvt_kernel<<<2048, 256, 0, stream>>>(x, x_bf, (long)T_TOK * HIDDEN);
  transpose_cvt_kernel<float><<<dim3(QKV_N / 32, HIDDEN / 32), dim3(32, 8), 0, stream>>>(
      wqkv, wqkv_t, HIDDEN, QKV_N);
  transpose_cvt_kernel<float><<<dim3(HIDDEN / 32, HIDDEN / 32), dim3(32, 8), 0, stream>>>(
      wo, wo_t, HIDDEN, HIDDEN);
  // 1) QKV projection (WMMA bf16, TN)
  gemm_tn_kernel<<<dim3(T_TOK / 64, QKV_N / 256), 256, 0, stream>>>(
      x_bf, wqkv_t, qkv, T_TOK, QKV_N, HIDDEN);
  // 2) RoPE Q -> bf16
  prep_q_kernel<<<(T_TOK * NH * (HD / 2) + 255) / 256, 256, 0, stream>>>(
      qkv, q_bf, seqstarts, start_pos);
  // 3) Gather + RoPE K / gather V -> bf16
  prep_kv_kernel<<<(TOTAL_KV * NKVH * (HD / 2) + 255) / 256, 256, 0, stream>>>(
      qkv, kv_cache, k_bf, v_bf, seqstarts, kvstarts, cachestarts, start_pos);
  // 3b) V -> V^T (contiguous-kv rows for P*V B-fragments)
  transpose_cvt_kernel<unsigned short><<<dim3((NKVH * HD) / 32, TOTAL_KV / 32), dim3(32, 8), 0, stream>>>(
      v_bf, v_t, TOTAL_KV, NKVH * HD);
  // 4) Flash attention (WMMA bf16)
  attn_kernel<<<(NH * (T_TOK / 16)) / 8, 256, 0, stream>>>(
      q_bf, k_bf, v_t, attn_bf, seqstarts, kvstarts, start_pos);
  // 5) Output projection (WMMA bf16, TN)
  gemm_tn_kernel<<<dim3(T_TOK / 64, HIDDEN / 256), 256, 0, stream>>>(
      attn_bf, wo_t, (float*)d_out, T_TOK, HIDDEN, HIDDEN);
}
